// PoincareLinear_27685359190128
// MI455X (gfx1250) — matrix operand order repaired and checked by reading, not compiled, not run on hardware
//
#include <hip/hip_runtime.h>
#include <hip/hip_bf16.h>
#include <math.h>

// Poincare MLR linear (HNN++) for MI455X / gfx1250.
// Core: [8192x4096]x[4096x4096] GEMM (275 GFLOP) via v_wmma_f32_16x16x32_f16.
// B tiles stream L2->LDS with global_load_async_to_lds_b128 (ASYNCcnt,
// triple-buffered); A fragments load straight from global in the ISA A-layout.
// All reductions deterministic (fixed-order trees, no float atomics).

#define IN_DIM   4096
#define OUT_DIM  4096
#define N_TOKENS 8192
#define EPSF     1e-15f
#define BK       32      // WMMA K per step
#define BUFB     10240   // bytes per LDS B buffer: 128 rows * 80B
#define NBUF     3

typedef __attribute__((ext_vector_type(16))) _Float16 v16h;
typedef __attribute__((ext_vector_type(8)))  float    v8f;

union Frag { uint4 u[2]; v16h h; };

// ---------------------------------------------------------------- kernel 1a:
// partial column sum-of-squares of weight_v (deterministic 2-pass reduction)
__global__ __launch_bounds__(256) void colnorm_partial(
    const float* __restrict__ w, float* __restrict__ part) {
  int o  = blockIdx.x * 256 + threadIdx.x;   // column (coalesced across lanes)
  int i0 = blockIdx.y * 256;                 // row chunk
  float s = 0.f;
#pragma unroll 8
  for (int i = 0; i < 256; ++i) {
    float v = w[(size_t)(i0 + i) * OUT_DIM + o];
    s = fmaf(v, v, s);
  }
  part[blockIdx.y * OUT_DIM + o] = s;
}

// ---------------------------------------------------------------- kernel 1b:
__global__ __launch_bounds__(256) void colnorm_finish(
    const float* __restrict__ part, float* __restrict__ invn) {
  int o = blockIdx.x * 256 + threadIdx.x;
  float s = 0.f;
#pragma unroll
  for (int b = 0; b < 16; ++b) s += part[b * OUT_DIM + o];
  invn[o] = 1.f / fmaxf(sqrtf(s), EPSF);     // 2nd normalization is idempotent
}

// ---------------------------------------------------------------- kernel 2:
// normalize weight columns, convert to f16, transpose to wT[OUT][IN] so a
// WMMA B-fragment (a column of B = 32 consecutive K) is contiguous in memory.
__global__ __launch_bounds__(256) void wprep(
    const float* __restrict__ w, const float* __restrict__ invn,
    _Float16* __restrict__ wT) {
  __shared__ _Float16 tile[64][72];
  int iBase = blockIdx.y * 64, oBase = blockIdx.x * 64;
  int tx = threadIdx.x & 63, ty = threadIdx.x >> 6;
#pragma unroll
  for (int r = 0; r < 16; ++r) {
    int i = ty + r * 4;
    float v = w[(size_t)(iBase + i) * OUT_DIM + oBase + tx] * invn[oBase + tx];
    tile[i][tx] = (_Float16)v;
  }
  __syncthreads();
#pragma unroll
  for (int r = 0; r < 16; ++r) {
    int o = ty + r * 4;
    wT[(size_t)(oBase + o) * IN_DIM + iBase + tx] = tile[tx][o];
  }
}

// ---------------------------------------------------------------- kernel 3:
// x -> f16 scaled by sqrt(c), and per-row cx2 = c*sum(x^2) (exact f32).
__global__ __launch_bounds__(256) void xprep(
    const float* __restrict__ x, const float* __restrict__ cptr,
    _Float16* __restrict__ xh, float* __restrict__ cx2) {
  __shared__ float red[256];
  int row = blockIdx.x;
  float c  = cptr[0];
  float rc = sqrtf(c);
  const float* xr = x  + (size_t)row * IN_DIM;
  _Float16*    xo = xh + (size_t)row * IN_DIM;
  float s = 0.f;
  for (int j = threadIdx.x; j < IN_DIM; j += 256) {
    float v = xr[j];
    s = fmaf(v, v, s);
    xo[j] = (_Float16)(rc * v);
  }
  red[threadIdx.x] = s;
  __syncthreads();
  for (int off = 128; off > 0; off >>= 1) {
    if (threadIdx.x < off) red[threadIdx.x] += red[threadIdx.x + off];
    __syncthreads();
  }
  if (threadIdx.x == 0) cx2[row] = c * red[0];
}

// ---------------------------------------------------------------- kernel 4:
// WMMA GEMM + fused Poincare MLR epilogue.
// 256 threads = 8 waves; block tile 256(M) x 128(N).
// Wave w: rows [blockRow + 32w, +32) as two 16-row A fragments -> 16 accums;
// each B fragment is reused twice (1 WMMA : 1 ds_load_b128).
// B tile (32K x 128N) streamed via global_load_async_to_lds_b128, 3 buffers.
__global__ __launch_bounds__(256) void poincare_gemm(
    const _Float16* __restrict__ xh, const _Float16* __restrict__ wT,
    const float* __restrict__ cx2,  const float* __restrict__ g,
    const float* __restrict__ bias, const float* __restrict__ cptr,
    float* __restrict__ out) {
  // B rows padded to 80B stride -> conflict-free b128 fragment reads.
  __shared__ _Float16 ldsB[NBUF][BUFB / 2];

  const int t    = threadIdx.x;
  const int wave = t >> 5, lane = t & 31;
  const int half = lane >> 4, l16 = lane & 15;
  const int rowBase = blockIdx.y * 256 + wave * 32;
  const int colBase = blockIdx.x * 128;

  // cooperative async B-tile load: 512 16B chunks (128 n-rows x 4), 2/thread
  const int n0 = t >> 2, q = t & 3, n1 = n0 + 64;
  const _Float16* bsrc0 = wT + (size_t)(colBase + n0) * IN_DIM + q * 8;
  const _Float16* bsrc1 = wT + (size_t)(colBase + n1) * IN_DIM + q * 8;
  const unsigned ldsBase = (unsigned)(uintptr_t)(&ldsB[0][0]); // flat->LDS off
  const unsigned bo0 = (unsigned)(n0 * 80 + q * 16);
  const unsigned bo1 = (unsigned)(n1 * 80 + q * 16);

  // A source: lane = row M=l16; ISA 16-bit A layout per 32-wide K step:
  // halfs [8*half .. +7] and [16+8*half .. +7].
  const _Float16* arow0 = xh + (size_t)(rowBase + l16)      * IN_DIM + 8 * half;
  const _Float16* arow1 = xh + (size_t)(rowBase + 16 + l16) * IN_DIM + 8 * half;

  v8f acc[2][8];
#pragma unroll
  for (int mi = 0; mi < 2; ++mi)
#pragma unroll
    for (int i = 0; i < 8; ++i)
      acc[mi][i] = (v8f){0.f,0.f,0.f,0.f,0.f,0.f,0.f,0.f};

#define ISSUE_TILE(TILE)                                                      \
  do {                                                                        \
    unsigned _dst = ldsBase + (unsigned)((TILE) % NBUF) * BUFB;               \
    unsigned long long _a0 =                                                  \
        (unsigned long long)(uintptr_t)(bsrc0 + (size_t)(TILE) * BK);         \
    unsigned long long _a1 =                                                  \
        (unsigned long long)(uintptr_t)(bsrc1 + (size_t)(TILE) * BK);         \
    asm volatile("global_load_async_to_lds_b128 %0, %1, off"                  \
                 :: "v"(_dst + bo0), "v"(_a0) : "memory");                    \
    asm volatile("global_load_async_to_lds_b128 %0, %1, off"                  \
                 :: "v"(_dst + bo1), "v"(_a1) : "memory");                    \
  } while (0)

  const int KSTEPS = IN_DIM / BK;            // 128
  ISSUE_TILE(0);
  ISSUE_TILE(1);

  for (int kb = 0; kb < KSTEPS; ++kb) {
    // my tile-kb chunks landed (tile kb+1 may still be in flight: 2 loads)
    if (kb + 1 < KSTEPS) asm volatile("s_wait_asynccnt 0x2" ::: "memory");
    else                 asm volatile("s_wait_asynccnt 0x0" ::: "memory");
    __syncthreads();     // everyone's tile kb in LDS; everyone done with kb-1
    if (kb + 2 < KSTEPS) ISSUE_TILE(kb + 2); // reuses buffer last read at kb-1

    Frag fa0, fa1;
    fa0.u[0] = *(const uint4*)(arow0 + (size_t)kb * BK);
    fa0.u[1] = *(const uint4*)(arow0 + (size_t)kb * BK + 16);
    fa1.u[0] = *(const uint4*)(arow1 + (size_t)kb * BK);
    fa1.u[1] = *(const uint4*)(arow1 + (size_t)kb * BK + 16);

    const char* cbuf = (const char*)&ldsB[0][0] + (size_t)(kb % NBUF) * BUFB;
#pragma unroll
    for (int ti = 0; ti < 8; ++ti) {         // 8 N-tiles
      Frag fb;                                // B: lane = column n, K 8h..+7 and 16+8h..+7
      const char* p = cbuf + (ti * 16 + l16) * 80 + 16 * half;
      fb.u[0] = *(const uint4*)p;
      fb.u[1] = *(const uint4*)(p + 32);
      acc[0][ti] = __builtin_amdgcn_wmma_f32_16x16x32_f16(
          false, fa0.h, false, fb.h, (short)0, acc[0][ti], false, false);
      acc[1][ti] = __builtin_amdgcn_wmma_f32_16x16x32_f16(
          false, fa1.h, false, fb.h, (short)0, acc[1][ti], false, false);
    }
  }
#undef ISSUE_TILE

  // -------- fused Poincare MLR epilogue --------
  float c   = cptr[0];
  float rcI = sqrtf(c);                      // rc inside the MLR
  float rcO = fmaxf(rcI, EPSF);              // rc for outer sinh/scale
  float cxv[2][8];
#pragma unroll
  for (int mi = 0; mi < 2; ++mi)
#pragma unroll
    for (int v = 0; v < 8; ++v)
      cxv[mi][v] = cx2[rowBase + mi * 16 + v + 8 * half];

#pragma unroll
  for (int ti = 0; ti < 8; ++ti) {
    int   n    = colBase + ti * 16 + l16;
    float drcr = 2.f * rcI * bias[n];
    float ch = coshf(drcr), sh = sinhf(drcr);
    float ysc = 2.f * g[n] / rcI;            // 2*z_norm/rc
#pragma unroll
    for (int mi = 0; mi < 2; ++mi) {
#pragma unroll
      for (int v = 0; v < 8; ++v) {
        float a   = acc[mi][ti][v];          // dot(rc*x, z_unit)
        float num = 2.f * a * ch - (1.f + cxv[mi][v]) * sh;
        float den = fmaxf(1.f - cxv[mi][v], EPSF);
        float y   = ysc * asinhf(num / den);
        if (isnan(y) || isinf(y)) y = 0.f;
        y = sinhf(fminf(fmaxf(rcO * y, -20.f), 20.f)) / rcO;
        if (isnan(y) || isinf(y)) y = 0.f;
        out[(size_t)(rowBase + mi * 16 + v + 8 * half) * OUT_DIM + n] = y;
      }
    }
  }
}

// ---------------------------------------------------------------- kernel 5:
// per-row yn2 reduction + final projection scale, in place on d_out.
__global__ __launch_bounds__(256) void rowfinal(
    float* __restrict__ out, const float* __restrict__ cptr) {
  __shared__ float red[256];
  float  c   = cptr[0];
  float* row = out + (size_t)blockIdx.x * OUT_DIM;
  float  s   = 0.f;
  for (int j = threadIdx.x; j < OUT_DIM; j += 256) {
    float v = row[j];
    s = fmaf(v, v, s);
  }
  red[threadIdx.x] = s;
  __syncthreads();
  for (int off = 128; off > 0; off >>= 1) {
    if (threadIdx.x < off) red[threadIdx.x] += red[threadIdx.x + off];
    __syncthreads();
  }
  float yn2   = fminf(fmaxf(red[0], 0.f), 1e15f);
  float denom = fmaxf(1.f + sqrtf(1.f + c * yn2), EPSF);
  float inv   = 1.f / denom;
  for (int j = threadIdx.x; j < OUT_DIM; j += 256) {
    float v = row[j] * inv;
    if (isnan(v) || isinf(v)) v = 0.f;
    row[j] = v;
  }
}

// ----------------------------------------------------------------------------
extern "C" void kernel_launch(void* const* d_in, const int* in_sizes, int n_in,
                              void* d_out, int out_size, void* d_ws, size_t ws_size,
                              hipStream_t stream) {
  const float* x  = (const float*)d_in[0];   // [N_TOKENS, IN_DIM]
  const float* wg = (const float*)d_in[1];   // [OUT_DIM]
  const float* wv = (const float*)d_in[2];   // [IN_DIM, OUT_DIM]
  const float* bs = (const float*)d_in[3];   // [OUT_DIM]
  const float* cp = (const float*)d_in[4];   // [1]
  float* out = (float*)d_out;

  // workspace layout (~96.3 MiB)
  char* ws = (char*)d_ws;
  _Float16* wT = (_Float16*)ws;                                           // 32 MiB
  _Float16* xh = (_Float16*)(ws + (size_t)IN_DIM * OUT_DIM * 2);          // 64 MiB
  float* cx2   = (float*)(ws + (size_t)IN_DIM * OUT_DIM * 2
                             + (size_t)N_TOKENS * IN_DIM * 2);
  float* invn  = cx2 + N_TOKENS;
  float* part  = invn + OUT_DIM;                                          // 16 x OUT_DIM

  colnorm_partial<<<dim3(OUT_DIM / 256, 16), 256, 0, stream>>>(wv, part);
  colnorm_finish <<<dim3(OUT_DIM / 256),     256, 0, stream>>>(part, invn);
  wprep          <<<dim3(OUT_DIM / 64, IN_DIM / 64), 256, 0, stream>>>(wv, invn, wT);
  xprep          <<<dim3(N_TOKENS),          256, 0, stream>>>(x, cp, xh, cx2);
  poincare_gemm  <<<dim3(OUT_DIM / 128, N_TOKENS / 256), 256, 0, stream>>>(
      xh, wT, cx2, wg, bs, cp, out);
  rowfinal       <<<dim3(N_TOKENS),          256, 0, stream>>>(out, cp);
}
